// ScaledDotProductAttention_35871566856662
// MI455X (gfx1250) — matrix operand order repaired and checked by reading, not compiled, not run on hardware
//
#include <hip/hip_runtime.h>
#include <math.h>

// Problem constants (from reference setup_inputs)
constexpr int B  = 16;
constexpr int NH = 8;
constexpr int T  = 2048;
constexpr int F  = 64;
constexpr float INV_TEMP = 0.125f;   // 1/8 : exact power of two -> fold into Q
constexpr float NEG_BIG  = -1.0e30f; // finite -inf sentinel (branchless softmax)

// LDS layout constants
constexpr int KROW_STRIDE = 72;      // padded bf16 row stride: conflict-free b128 reads
constexpr int KC          = 512;     // key-chunk length for PV kernel
constexpr int VT_STRIDE   = KC + 16; // padded bf16 key stride for transposed V (528)

typedef __attribute__((ext_vector_type(16))) __bf16 v16bf;
typedef __attribute__((ext_vector_type(8)))  __bf16 v8bf;
typedef __attribute__((ext_vector_type(8)))  float  v8f;

struct alignas(8) bf4 { __bf16 v[4]; };
struct alignas(8) us4 { unsigned short v[4]; };

// Per-lane K-index map for the 16-bit A matrix (16x32, MxK), wave32:
// element e: K = (lane/16)*16 + e (contiguous order)
__device__ __forceinline__ int kmapA(int e, int g) {
    return (g << 4) + e;
}

__device__ __forceinline__ v16bf cat16(v8bf lo, v8bf hi) {
    return __builtin_shufflevector(lo, hi, 0, 1, 2, 3, 4, 5, 6, 7,
                                            8, 9, 10, 11, 12, 13, 14, 15);
}

// ---------------------------------------------------------------------------
// Kernel 1: scores + masked softmax, writes normalized attn (fp32) once.
// One wave per 16-query tile; block = 8 waves = 128 query rows.
// LDS (exactly 320 KB, the CDNA5 WGP maximum):
//   - K[b] as bf16, 2048 x 72  (288 KB), conflict-free b128 fragment reads
//   - bit-packed mask, uint16 per (wave, column) (32 KB): bit r = row masked
// Flash-style: pass A keeps per-row (max, sumexp) online and branchless;
// pass B recomputes scores via WMMA (compute is free vs. the 256 MB attn
// write) and stores probabilities exactly once. Hot loops touch global
// memory only for the final attn store.
// ---------------------------------------------------------------------------
__global__ void attn_softmax_wmma(const float* __restrict__ q,
                                  const float* __restrict__ k,
                                  const unsigned char* __restrict__ mask,
                                  float* __restrict__ attn) {
    extern __shared__ __bf16 kb[];   // [T][KROW_STRIDE] bf16, then mask bits
    unsigned short* mb16 = (unsigned short*)(kb + T * KROW_STRIDE); // [8][T]

    const int tid   = threadIdx.x;
    const int lane  = tid & 31;
    const int w     = tid >> 5;
    const int row16 = lane & 15;     // N coordinate / A-row coordinate
    const int g     = lane >> 4;     // lane half
    const int b     = blockIdx.x >> 4;
    const int db    = blockIdx.x & 15;
    const int qb0   = db * 128;      // block's first query row
    const int qbase = qb0 + w * 16;  // this wave's first query row

    // ---- Stage K[b] -> LDS as bf16 (float4 in, packed 4x bf16 out) ----
    const float* kg = k + (size_t)b * T * F;
    for (int it = 0; it < (T * F / 4) / 256; ++it) {
        int idx4 = tid + it * 256;
        int row  = idx4 >> 4;            // 16 float4 per 64-feature row
        int fq   = (idx4 & 15) << 2;
        float4 kv = *(const float4*)(kg + row * F + fq);
        bf4 pk;
        pk.v[0] = (__bf16)kv.x; pk.v[1] = (__bf16)kv.y;
        pk.v[2] = (__bf16)kv.z; pk.v[3] = (__bf16)kv.w;
        *(bf4*)&kb[row * KROW_STRIDE + fq] = pk;
    }

    // ---- Bit-pack this block's 128 mask rows into LDS (coalesced) ----
    // mb16[wv][col] bit r  <=>  mask[b][qb0 + wv*16 + r][col]
    const unsigned char* mrow = mask + (size_t)b * T * T;
    for (int it = 0; it < 16; ++it) {
        int o    = tid + it * 256;       // 0..4095
        int wv   = o >> 9;               // wave group 0..7
        int col0 = (o & 511) << 2;       // 4 columns at a time
        us4 pk = {{0, 0, 0, 0}};
        for (int r = 0; r < 16; ++r) {
            unsigned int m4 = *(const unsigned int*)(
                mrow + (size_t)(qb0 + wv * 16 + r) * T + col0);
            pk.v[0] |= (unsigned short)(((m4 & 0x000000ffu) ? 1u : 0u) << r);
            pk.v[1] |= (unsigned short)(((m4 & 0x0000ff00u) ? 1u : 0u) << r);
            pk.v[2] |= (unsigned short)(((m4 & 0x00ff0000u) ? 1u : 0u) << r);
            pk.v[3] |= (unsigned short)(((m4 & 0xff000000u) ? 1u : 0u) << r);
        }
        *(us4*)&mb16[wv * T + col0] = pk;
    }
    __syncthreads();

    // ---- Load Q A-fragments, pre-scaled by 1/8 (exact in bf16) ----
    v16bf a0, a1;
    const float* qrow = q + ((size_t)b * T + qbase + row16) * F;
    for (int p = 0; p < 8; ++p) {
        int K = kmapA(2 * p, g);
        float2 lo = *(const float2*)(qrow + K);
        float2 hi = *(const float2*)(qrow + 32 + K);
        a0[2 * p]     = (__bf16)(lo.x * INV_TEMP);
        a0[2 * p + 1] = (__bf16)(lo.y * INV_TEMP);
        a1[2 * p]     = (__bf16)(hi.x * INV_TEMP);
        a1[2 * p + 1] = (__bf16)(hi.y * INV_TEMP);
    }

    float m[8], s[8];
    for (int j = 0; j < 8; ++j) { m[j] = NEG_BIG; s[j] = 0.0f; }

    // ---- Pass A: branchless online (max, sumexp) over 128 key tiles ----
    for (int kt = 0; kt < T / 16; ++kt) {
        const __bf16* kp = &kb[(kt * 16 + row16) * KROW_STRIDE + g * 16];
        v16bf b0 = cat16(*(const v8bf*)kp,        *(const v8bf*)(kp + 8));
        v16bf b1 = cat16(*(const v8bf*)(kp + 32), *(const v8bf*)(kp + 40));

        unsigned int mbits = mb16[w * T + kt * 16 + row16];

        v8f c = {};
        c = __builtin_amdgcn_wmma_f32_16x16x32_bf16(false, a0, false, b0,
                                                    (short)0, c, false, false);
        c = __builtin_amdgcn_wmma_f32_16x16x32_bf16(false, a1, false, b1,
                                                    (short)0, c, false, false);

        for (int j = 0; j < 8; ++j) {
            bool mk  = (mbits >> (j + 8 * g)) & 1u;
            float sc = mk ? NEG_BIG : c[j];
            float nm = fmaxf(m[j], sc);
            s[j] = s[j] * __expf(m[j] - nm) + __expf(sc - nm);
            m[j] = nm;
        }
    }

    // ---- Cross-lane (butterfly within 16-lane halves), branchless ----
    for (int xm = 1; xm <= 8; xm <<= 1) {
        for (int j = 0; j < 8; ++j) {
            float om = __shfl_xor(m[j], xm, 32);
            float os = __shfl_xor(s[j], xm, 32);
            float nm = fmaxf(m[j], om);
            s[j] = s[j] * __expf(m[j] - nm) + os * __expf(om - nm);
            m[j] = nm;
        }
    }
    float inv[8];
    for (int j = 0; j < 8; ++j) inv[j] = 1.0f / s[j];

    // ---- Pass B: recompute scores, write normalized probabilities once ----
    for (int kt = 0; kt < T / 16; ++kt) {
        const __bf16* kp = &kb[(kt * 16 + row16) * KROW_STRIDE + g * 16];
        v16bf b0 = cat16(*(const v8bf*)kp,        *(const v8bf*)(kp + 8));
        v16bf b1 = cat16(*(const v8bf*)(kp + 32), *(const v8bf*)(kp + 40));

        unsigned int mbits = mb16[w * T + kt * 16 + row16];

        v8f c = {};
        c = __builtin_amdgcn_wmma_f32_16x16x32_bf16(false, a0, false, b0,
                                                    (short)0, c, false, false);
        c = __builtin_amdgcn_wmma_f32_16x16x32_bf16(false, a1, false, b1,
                                                    (short)0, c, false, false);

        const int col = kt * 16 + row16;
        for (int j = 0; j < 8; ++j) {
            bool mk = (mbits >> (j + 8 * g)) & 1u;
            float e = __expf(c[j] - m[j]) * inv[j];
            int r = qbase + j + 8 * g;
            attn[((size_t)b * T + r) * T + col] = mk ? 0.0f : e;
        }
    }
}

// ---------------------------------------------------------------------------
// Kernel 2: output[b,n] = attn[b] (2048x2048) x v[b,n] (2048x64), bf16 WMMA.
// Block = 8 waves = 128 query rows for one (b, n). V staged transposed in LDS
// in KC=512 key chunks so B-fragments are contiguous 32-byte LDS reads.
// attn rows are freshly written -> served from the 192 MB L2.
// Per-wave accumulator: 16x64 fp32 (4 C tiles = 32 VGPRs).
// ---------------------------------------------------------------------------
__global__ void attn_v_wmma(const float* __restrict__ v,
                            const float* __restrict__ attn,
                            float* __restrict__ out) {
    extern __shared__ __bf16 vt[];   // [F][VT_STRIDE]  (transposed V chunk)

    const int tid   = threadIdx.x;
    const int lane  = tid & 31;
    const int w     = tid >> 5;
    const int row16 = lane & 15;
    const int g     = lane >> 4;
    const int bx    = blockIdx.x;
    const int b     = bx >> 7;        // 8 heads * 16 dblocks per batch
    const int n     = (bx >> 4) & 7;
    const int dblk  = bx & 15;
    const int qbase = dblk * 128 + w * 16;

    const float* vbase = v + ((size_t)b * NH + n) * T * F;
    const float* prow  = attn + ((size_t)b * T + qbase + row16) * T;

    v8f acc[4] = {};

    for (int chunk = 0; chunk < T / KC; ++chunk) {
        __syncthreads();
        // ---- Stage V chunk transposed into LDS (bf16) ----
        for (int it = 0; it < (KC * F / 4) / 256; ++it) {
            int idx4 = tid + it * 256;
            int key  = idx4 >> 4;
            int fq   = (idx4 & 15) << 2;
            float4 vv = *(const float4*)(vbase +
                         (size_t)(chunk * KC + key) * F + fq);
            vt[(fq + 0) * VT_STRIDE + key] = (__bf16)vv.x;
            vt[(fq + 1) * VT_STRIDE + key] = (__bf16)vv.y;
            vt[(fq + 2) * VT_STRIDE + key] = (__bf16)vv.z;
            vt[(fq + 3) * VT_STRIDE + key] = (__bf16)vv.w;
        }
        __syncthreads();

        for (int kt = 0; kt < KC / 32; ++kt) {
            // A fragment: 16x32 slice of attn probabilities, fp32 -> bf16
            int keyb = chunk * KC + kt * 32;
            __builtin_prefetch(prow + keyb + 64, 0, 0);
            v16bf a;
            for (int p = 0; p < 8; ++p) {
                int K = kmapA(2 * p, g);
                float2 pp = *(const float2*)(prow + keyb + K);
                a[2 * p]     = (__bf16)pp.x;
                a[2 * p + 1] = (__bf16)pp.y;
            }
            // B fragments: V^T rows are contiguous along the key dimension
            for (int nt = 0; nt < 4; ++nt) {
                const __bf16* vp = &vt[(nt * 16 + row16) * VT_STRIDE +
                                       kt * 32 + g * 16];
                v16bf bf = cat16(*(const v8bf*)vp, *(const v8bf*)(vp + 8));
                acc[nt] = __builtin_amdgcn_wmma_f32_16x16x32_bf16(
                    false, a, false, bf, (short)0, acc[nt], false, false);
            }
        }
    }

    // ---- Store 16x64 fp32 output tile (coalesced across lanes) ----
    float* obase = out + ((size_t)b * NH + n) * T * F;
    for (int nt = 0; nt < 4; ++nt) {
        for (int j = 0; j < 8; ++j) {
            obase[(size_t)(qbase + j + 8 * g) * F + nt * 16 + row16] =
                acc[nt][j];
        }
    }
}

// ---------------------------------------------------------------------------
extern "C" void kernel_launch(void* const* d_in, const int* in_sizes, int n_in,
                              void* d_out, int out_size, void* d_ws,
                              size_t ws_size, hipStream_t stream) {
    const float* q = (const float*)d_in[0];
    const float* k = (const float*)d_in[1];
    const float* v = (const float*)d_in[2];
    const unsigned char* mask = (const unsigned char*)d_in[3]; // bool array

    float* out  = (float*)d_out;                      // (b, n, t, f)
    float* attn = out + (size_t)B * NH * T * F;       // (b, t, t)

    // Kernel 1: 16 batches x 16 row-blocks; 288 KB K + 32 KB mask bits
    // = exactly the 320 KB CDNA5 WGP LDS capacity.
    size_t smem1 = (size_t)T * KROW_STRIDE * sizeof(__bf16)
                 + (size_t)8 * T * sizeof(unsigned short);
    attn_softmax_wmma<<<dim3(B * 16), dim3(256), smem1, stream>>>(q, k, mask,
                                                                  attn);

    // Kernel 2: 16 batches x 8 heads x 16 row-blocks; 66 KB dynamic LDS.
    size_t smem2 = (size_t)F * VT_STRIDE * sizeof(__bf16);
    attn_v_wmma<<<dim3(B * NH * 16), dim3(256), smem2, stream>>>(v, attn, out);
}
